// BehaviorModel_20160576487889
// MI455X (gfx1250) — compile-verified, matrix operand order checked by reading, not run on hardware
//
#include <hip/hip_runtime.h>
#include <math.h>

// Problem constants (from reference)
#define H_   256
#define K_   128
#define B_   512
#define T_   64
#define TOUT 512   // output_length = pose_sequence.shape[0] = 512

typedef _Float16 v16h __attribute__((ext_vector_type(16)));
typedef float    v8f  __attribute__((ext_vector_type(8)));
typedef _Float16 h2v  __attribute__((ext_vector_type(2)));
typedef unsigned int v4u __attribute__((ext_vector_type(4)));
typedef int          v8i __attribute__((ext_vector_type(8)));
typedef int          v4i __attribute__((ext_vector_type(4)));

// --- K index pattern for 16-bit WMMA fragments (ISA 7.12.2, 16-bit 16x32 A layout,
// mirrored for the KxN B operand). pair p (0..7), lane-group g (0/1).
__device__ __forceinline__ int kpair(int p, int g) {
    return (p < 4 ? 2 * p : 16 + 2 * (p - 4)) + 8 * g;
}

// Load a 16x32 f16 A fragment from a row-major f16 matrix (rows = batch, ld = row stride).
// lane L: row m = L&15, group g = L>>4 ; pairs coalesce into two 16B loads per lane.
__device__ __forceinline__ v16h load_a(const _Float16* __restrict__ base, int ld, int lane) {
    const int m = lane & 15, g = lane >> 4;
    const _Float16* p = base + (size_t)m * ld;
    v16h a;
#pragma unroll
    for (int pr = 0; pr < 8; ++pr) {
        const int k = kpair(pr, g);
        h2v t = *(const h2v*)(p + k);
        a[2 * pr]     = t.x;
        a[2 * pr + 1] = t.y;
    }
    return a;
}

// ---------------------------------------------------------------------------
// Tensor Data Mover: async DMA of a 2D f16 tile (rows x rowElems, row stride
// strideElems) from global memory into LDS at lds_off. D# built per CDNA5 ISA
// §10.8 (group0: count/lds_addr/global_addr/type=2; group1: data_size=2B,
// tensor dims, tile dims, dim0 stride). Tracked by TENSORcnt.
__device__ __forceinline__ void tdm_load_2d(const void* gaddr, unsigned lds_off,
                                            unsigned rowElems, unsigned rows,
                                            unsigned strideElems, unsigned tensorRows) {
    unsigned long long ga = (unsigned long long)gaddr;
    v4u g0;
    g0[0] = 1u;                                            // count=1, user mode
    g0[1] = lds_off;                                       // lds_addr (bytes)
    g0[2] = (unsigned)(ga & 0xffffffffu);                  // global_addr[31:0]
    g0[3] = (unsigned)((ga >> 32) & 0x1ffffffu)            // global_addr[56:32]
          | (2u << 30);                                    // type = 2 ("image")
    v8i g1;
    g1[0] = (int)(1u << 16);                               // data_size = 1 -> 2 bytes
    g1[1] = (int)((rowElems & 0xffffu) << 16);             // tensor_dim0[15:0]
    g1[2] = (int)(((rowElems >> 16) & 0xffffu)             // tensor_dim0[31:16]
          | ((tensorRows & 0xffffu) << 16));               // tensor_dim1[15:0]
    g1[3] = (int)(((tensorRows >> 16) & 0xffffu)           // tensor_dim1[31:16]
          | ((rowElems & 0xffffu) << 16));                 // tile_dim0 = full row
    g1[4] = (int)(rows & 0xffffu);                         // tile_dim1 (tile_dim2=0)
    g1[5] = (int)strideElems;                              // tensor_dim0_stride[31:0]
    g1[6] = 0;                                             // stride[47:32], d1s[15:0]
    g1[7] = 0;
    v4i z4 = {0, 0, 0, 0};
#if __clang_major__ >= 23
    v8i z8 = {0, 0, 0, 0, 0, 0, 0, 0};
    __builtin_amdgcn_tensor_load_to_lds(g0, g1, z4, z4, z8, 0);
#else
    __builtin_amdgcn_tensor_load_to_lds(g0, g1, z4, z4, 0);
#endif
}

// ---------------------------------------------------------------------------
// Packing: fp32 weight (R rows = output neurons, C cols = reduction dim) ->
// f16 tiles [R/16][C/32], each tile = 32 lanes x 16 halves laid out exactly as
// the WMMA B fragment, so the per-wave fragment load is contiguous (32B/lane).
__global__ void pack_weight(const float* __restrict__ src, _Float16* __restrict__ dst,
                            int R, int C) {
    int gid = blockIdx.x * blockDim.x + threadIdx.x;
    int n = R * C;
    if (gid >= n) return;
    int tile   = gid >> 9;        // /512
    int within = gid & 511;
    int lane   = within >> 4;
    int hh     = within & 15;
    int ntk    = C >> 5;          // tiles along K
    int tr     = tile / ntk;
    int tk     = tile % ntk;
    int r      = tr * 16 + (lane & 15);
    int k      = tk * 32 + kpair(hh >> 1, lane >> 4) + (hh & 1);
    dst[gid] = (_Float16)src[(size_t)r * C + k];
}

// pose_sequence (B,T,K) fp32 -> time-major f16 [T][B][K]
__global__ void convert_x(const float* __restrict__ src, _Float16* __restrict__ dst) {
    int gid = blockIdx.x * blockDim.x + threadIdx.x;
    int n = B_ * T_ * K_;
    if (gid >= n) return;
    int k = gid % K_;
    int t = (gid / K_) % T_;
    int b = gid / (K_ * T_);
    dst[((size_t)t * B_ + b) * K_ + k] = (_Float16)src[gid];
}

__global__ void zero_f16(_Float16* __restrict__ p, int n) {
    int i = blockIdx.x * blockDim.x + threadIdx.x;
    if (i < n) p[i] = (_Float16)0.0f;
}

// ---------------------------------------------------------------------------
// One GRU layer, one time step. Block = one 16-row batch tile; its 16 waves
// each own one 16-col h tile and compute all three gates for it:
//   accr,accz accumulate [x|h] @ [Wih|Whh]^T for columns j and j+H (fused)
//   accin = x @ Wih_n^T  ;  acchn = h @ Whh_n^T   (kept separate for r*h_n)
// Wave 0 DMAs the shared x/h A-tiles into LDS via the Tensor Data Mover.
__global__ __launch_bounds__(512)
void gru_layer(const _Float16* __restrict__ x, int Kx,
               const _Float16* __restrict__ Wih_p, const _Float16* __restrict__ Whh_p,
               const float* __restrict__ bih, const float* __restrict__ bhh,
               const _Float16* __restrict__ h_in, _Float16* __restrict__ h_out) {
    __shared__ __align__(128) _Float16 sx[16 * 256];   // x tile (16 x Kx)
    __shared__ __align__(128) _Float16 sh[16 * 256];   // h tile (16 x 256)

    const int lane    = threadIdx.x & 31;
    const int colTile = threadIdx.x >> 5;   // 0..15 (H/16)
    const int rowTile = blockIdx.x;         // 0..31 (B/16)
    const int b0 = rowTile * 16;
    const int j0 = colTile * 16;

    if (threadIdx.x < 32) {  // wave 0 issues both TDM loads, waits TENSORcnt
        tdm_load_2d(x + (size_t)b0 * Kx, (unsigned)(size_t)&sx[0], (unsigned)Kx, 16u,
                    (unsigned)Kx, (unsigned)B_);
        tdm_load_2d(h_in + (size_t)b0 * H_, (unsigned)(size_t)&sh[0], (unsigned)H_, 16u,
                    (unsigned)H_, (unsigned)B_);
        __builtin_amdgcn_s_wait_tensorcnt(0);
    }
    __syncthreads();

    v8f accr = {}, accz = {}, accin = {}, acchn = {};

    // ---- x part: Wih has 3H=768 rows, C=Kx. Gate g tile-row = colTile + g*16.
    const int nkx = Kx >> 5;
    const _Float16* wl = Wih_p + (size_t)lane * 16;
    for (int kc = 0; kc < nkx; ++kc) {
        v16h a = load_a(sx + kc * 32, Kx, lane);
        size_t t0 = ((size_t)(colTile)*nkx + kc) * 512;
        size_t t1 = ((size_t)(colTile + 16) * nkx + kc) * 512;
        size_t t2 = ((size_t)(colTile + 32) * nkx + kc) * 512;
        v16h br = *(const v16h*)(wl + t0);
        v16h bz = *(const v16h*)(wl + t1);
        v16h bn = *(const v16h*)(wl + t2);
        if (kc + 1 < nkx) {  // prefetch next K-chunk's weight tiles
            __builtin_prefetch(wl + t0 + 512, 0, 3);
            __builtin_prefetch(wl + t1 + 512, 0, 3);
            __builtin_prefetch(wl + t2 + 512, 0, 3);
        }
        accr  = __builtin_amdgcn_wmma_f32_16x16x32_f16(false, a, false, br, (short)0, accr,  false, false);
        accz  = __builtin_amdgcn_wmma_f32_16x16x32_f16(false, a, false, bz, (short)0, accz,  false, false);
        accin = __builtin_amdgcn_wmma_f32_16x16x32_f16(false, a, false, bn, (short)0, accin, false, false);
    }

    // ---- h part: Whh has 768 rows, C=H=256
    const int nkh = H_ >> 5;  // 8
    const _Float16* wh = Whh_p + (size_t)lane * 16;
    for (int kc = 0; kc < nkh; ++kc) {
        v16h a = load_a(sh + kc * 32, H_, lane);
        size_t t0 = ((size_t)(colTile)*nkh + kc) * 512;
        size_t t1 = ((size_t)(colTile + 16) * nkh + kc) * 512;
        size_t t2 = ((size_t)(colTile + 32) * nkh + kc) * 512;
        v16h br = *(const v16h*)(wh + t0);
        v16h bz = *(const v16h*)(wh + t1);
        v16h bn = *(const v16h*)(wh + t2);
        if (kc + 1 < nkh) {
            __builtin_prefetch(wh + t0 + 512, 0, 3);
            __builtin_prefetch(wh + t1 + 512, 0, 3);
            __builtin_prefetch(wh + t2 + 512, 0, 3);
        }
        accr  = __builtin_amdgcn_wmma_f32_16x16x32_f16(false, a, false, br, (short)0, accr,  false, false);
        accz  = __builtin_amdgcn_wmma_f32_16x16x32_f16(false, a, false, bz, (short)0, accz,  false, false);
        acchn = __builtin_amdgcn_wmma_f32_16x16x32_f16(false, a, false, bn, (short)0, acchn, false, false);
    }

    // ---- pointwise GRU gate math. C/D layout: VGPR i -> M = i + 8*(lane>>4), N = lane&15
    const int g = lane >> 4;
    const int n = lane & 15;
    const int j = j0 + n;
    const float brz = bih[j] + bhh[j];
    const float bzz = bih[j + H_] + bhh[j + H_];
    const float bin = bih[j + 2 * H_];
    const float bhn = bhh[j + 2 * H_];
#pragma unroll
    for (int i = 0; i < 8; ++i) {
        const int m = i + 8 * g;                      // row within tile
        float r  = 1.0f / (1.0f + __expf(-(accr[i] + brz)));
        float z  = 1.0f / (1.0f + __expf(-(accz[i] + bzz)));
        float nn = tanhf(accin[i] + bin + r * (acchn[i] + bhn));
        float hp = (float)sh[m * H_ + j];             // h_prev from staged LDS tile
        h_out[(size_t)(b0 + m) * H_ + j] = (_Float16)((1.0f - z) * nn + z * hp);
    }
}

// ---------------------------------------------------------------------------
// Projection: out = top @ out_W^T + out_b ; writes fp32 to d_out[:, tstep, :]
// and the f16 copy used as next autoregressive input. Top tile staged via TDM.
__global__ __launch_bounds__(256)
void proj_kernel(const _Float16* __restrict__ top, const _Float16* __restrict__ Wp,
                 const float* __restrict__ ob, float* __restrict__ out, int tstep,
                 _Float16* __restrict__ xnext) {
    __shared__ __align__(128) _Float16 st[16 * 256];

    const int lane    = threadIdx.x & 31;
    const int colTile = threadIdx.x >> 5;   // 0..7 (K/16)
    const int rowTile = blockIdx.x;         // 0..31 (B/16)
    const int b0 = rowTile * 16;
    const int j0 = colTile * 16;

    if (threadIdx.x < 32) {
        tdm_load_2d(top + (size_t)b0 * H_, (unsigned)(size_t)&st[0], (unsigned)H_, 16u,
                    (unsigned)H_, (unsigned)B_);
        __builtin_amdgcn_s_wait_tensorcnt(0);
    }
    __syncthreads();

    v8f acc = {};
    const int nk = H_ >> 5;  // 8
    const _Float16* wl = Wp + (size_t)lane * 16;
    for (int kc = 0; kc < nk; ++kc) {
        v16h a = load_a(st + kc * 32, H_, lane);
        v16h b = *(const v16h*)(wl + ((size_t)colTile * nk + kc) * 512);
        acc = __builtin_amdgcn_wmma_f32_16x16x32_f16(false, a, false, b, (short)0, acc, false, false);
    }
    const int g = lane >> 4;
    const int n = lane & 15;
    const int j = j0 + n;
    const float bb = ob[j];
#pragma unroll
    for (int i = 0; i < 8; ++i) {
        const int b = b0 + i + 8 * g;
        float v = acc[i] + bb;
        out[((size_t)b * TOUT + tstep) * K_ + j] = v;
        xnext[(size_t)b * K_ + j] = (_Float16)v;
    }
}

// ---------------------------------------------------------------------------
extern "C" void kernel_launch(void* const* d_in, const int* in_sizes, int n_in,
                              void* d_out, int out_size, void* d_ws, size_t ws_size,
                              hipStream_t stream) {
    (void)in_sizes; (void)n_in; (void)out_size; (void)ws_size;

    const float* pose     = (const float*)d_in[0];
    const float* eWih0    = (const float*)d_in[1];
    const float* eWhh0    = (const float*)d_in[2];
    const float* ebih0    = (const float*)d_in[3];
    const float* ebhh0    = (const float*)d_in[4];
    const float* eWih1    = (const float*)d_in[5];
    const float* eWhh1    = (const float*)d_in[6];
    const float* ebih1    = (const float*)d_in[7];
    const float* ebhh1    = (const float*)d_in[8];
    const float* dWih0    = (const float*)d_in[9];
    const float* dWhh0    = (const float*)d_in[10];
    const float* dbih0    = (const float*)d_in[11];
    const float* dbhh0    = (const float*)d_in[12];
    const float* dWih1    = (const float*)d_in[13];
    const float* dWhh1    = (const float*)d_in[14];
    const float* dbih1    = (const float*)d_in[15];
    const float* dbhh1    = (const float*)d_in[16];
    const float* outW     = (const float*)d_in[17];
    const float* outb     = (const float*)d_in[18];
    float* out = (float*)d_out;

    // ---- workspace carve-up (all offsets 256B aligned)
    char* ws = (char*)d_ws;
    size_t off = 0;
    auto carve = [&](size_t halves) -> _Float16* {
        _Float16* p = (_Float16*)(ws + off);
        off += (halves * sizeof(_Float16) + 255) & ~(size_t)255;
        return p;
    };
    _Float16* x_all   = carve((size_t)T_ * B_ * K_);   // time-major pose f16
    _Float16* pEW0i   = carve(768 * 128);
    _Float16* pEW0h   = carve(768 * 256);
    _Float16* pEW1i   = carve(768 * 256);
    _Float16* pEW1h   = carve(768 * 256);
    _Float16* pDW0i   = carve(768 * 128);
    _Float16* pDW0h   = carve(768 * 256);
    _Float16* pDW1i   = carve(768 * 256);
    _Float16* pDW1h   = carve(768 * 256);
    _Float16* pOutW   = carve(128 * 256);
    _Float16* hbuf    = carve((size_t)4 * B_ * H_);    // [layer][pingpong][B][H]
    _Float16* x_dec   = carve((size_t)B_ * K_);        // autoregressive f16 input

    _Float16* hb[2][2] = {
        { hbuf,                      hbuf + (size_t)B_ * H_ },
        { hbuf + (size_t)2*B_*H_,    hbuf + (size_t)3*B_*H_ } };

    // ---- one-time prep (re-run every call: deterministic)
    {
        int n = B_ * T_ * K_;
        convert_x<<<(n + 255) / 256, 256, 0, stream>>>(pose, x_all);
    }
    auto pack = [&](const float* s, _Float16* d, int R, int C) {
        int n = R * C;
        pack_weight<<<(n + 255) / 256, 256, 0, stream>>>(s, d, R, C);
    };
    pack(eWih0, pEW0i, 768, 128);  pack(eWhh0, pEW0h, 768, 256);
    pack(eWih1, pEW1i, 768, 256);  pack(eWhh1, pEW1h, 768, 256);
    pack(dWih0, pDW0i, 768, 128);  pack(dWhh0, pDW0h, 768, 256);
    pack(dWih1, pDW1i, 768, 256);  pack(dWhh1, pDW1h, 768, 256);
    pack(outW,  pOutW, 128, 256);
    {
        int n = 4 * B_ * H_;
        zero_f16<<<(n + 255) / 256, 256, 0, stream>>>(hbuf, n);
    }

    // ---- recurrence: 32 blocks x 512 threads (16 wave32) per layer step
    int c0 = 0, c1 = 0;
    auto layer = [&](const _Float16* x, int Kx, const _Float16* Wi, const _Float16* Wh,
                     const float* bi, const float* bh, int l) -> const _Float16* {
        int& c = l ? c1 : c0;
        _Float16* hin  = hb[l][c];
        _Float16* hout = hb[l][c ^ 1];
        gru_layer<<<32, 512, 0, stream>>>(x, Kx, Wi, Wh, bi, bh, hin, hout);
        c ^= 1;
        return hout;
    };

    // encoder: 64 steps
    for (int t = 0; t < T_; ++t) {
        const _Float16* xs = x_all + (size_t)t * B_ * K_;
        const _Float16* h0 = layer(xs, K_, pEW0i, pEW0h, ebih0, ebhh0, 0);
        layer(h0, H_, pEW1i, pEW1h, ebih1, ebhh1, 1);
    }
    // decoder phase 1: teacher-forced on pose slices, 64 steps
    for (int t = 0; t < T_; ++t) {
        const _Float16* xs = x_all + (size_t)t * B_ * K_;
        const _Float16* h0 = layer(xs, K_, pDW0i, pDW0h, dbih0, dbhh0, 0);
        const _Float16* h1 = layer(h0, H_, pDW1i, pDW1h, dbih1, dbhh1, 1);
        proj_kernel<<<32, 256, 0, stream>>>(h1, pOutW, outb, out, t, x_dec);
    }
    // decoder phase 2: autoregressive, 448 steps
    for (int t = T_; t < TOUT; ++t) {
        const _Float16* h0 = layer(x_dec, K_, pDW0i, pDW0h, dbih0, dbhh0, 0);
        const _Float16* h1 = layer(h0, H_, pDW1i, pDW1h, dbih1, dbhh1, 1);
        proj_kernel<<<32, 256, 0, stream>>>(h1, pOutW, outb, out, t, x_dec);
    }
}
